// Attention_43456479101595
// MI455X (gfx1250) — compile-verified, matrix operand order checked by reading, not run on hardware
//
#include <hip/hip_runtime.h>
#include <hip/hip_bf16.h>
#include <math.h>

typedef __attribute__((ext_vector_type(16))) __bf16 v16bf;
typedef __attribute__((ext_vector_type(8)))  float  v8f;

#define HD   1024
#define SS   2048
#define BB   64
#define MBLK 64
#define LDSK (HD + 8)   // bf16 elements per LDS A row (row stride 2064B: 16B aligned, bank-spread)

#if __has_builtin(__builtin_amdgcn_tanhf)
#define TANHF(x) __builtin_amdgcn_tanhf(x)
#elif __has_builtin(__builtin_amdgcn_tanh_f32)
#define TANHF(x) __builtin_amdgcn_tanh_f32(x)
#else
#define TANHF(x) tanhf(x)
#endif

static __device__ __forceinline__ unsigned short f2bf(float x) {
    union { float f; unsigned u; } v; v.f = x;
    unsigned r = v.u + 0x7FFFu + ((v.u >> 16) & 1u);   // round-to-nearest-even
    return (unsigned short)(r >> 16);
}

// ---------------- Kernel 1: W1 [k][n] fp32 -> W1T [n][k] bf16 (tiled transpose) -----------
__global__ __launch_bounds__(256) void w1t_kernel(const float* __restrict__ w1,
                                                  unsigned short* __restrict__ w1t) {
    __shared__ float tile[32][33];
    const int kb = blockIdx.x * 32;   // k tile base
    const int nb = blockIdx.y * 32;   // n tile base
    const int tx = threadIdx.x & 31;
    const int ty = threadIdx.x >> 5;  // 8 rows at a time
#pragma unroll
    for (int r = ty; r < 32; r += 8)
        tile[r][tx] = w1[(size_t)(kb + r) * HD + (nb + tx)];
    __syncthreads();
#pragma unroll
    for (int r = ty; r < 32; r += 8)
        w1t[(size_t)(nb + r) * HD + (kb + tx)] = f2bf(tile[tx][r]);
}

// ---------------- Kernel 2: dec_proj[b][n] = dh[b]@W2[:,n] + b2[n] ------------------------
__global__ __launch_bounds__(256) void decproj_kernel(const float* __restrict__ dh,
                                                      const float* __restrict__ w2,
                                                      const float* __restrict__ w2b,
                                                      float* __restrict__ dp) {
    const int idx = blockIdx.x * 256 + threadIdx.x;   // over B*H
    const int b   = idx >> 10;
    const int n   = idx & (HD - 1);
    const float* dhr = dh + (size_t)b * HD;
    float acc = 0.f;
#pragma unroll 8
    for (int k = 0; k < HD; ++k)
        acc = fmaf(dhr[k], w2[(size_t)k * HD + n], acc);
    dp[idx] = acc + w2b[n];
}

// ---------------- Kernel 3: fused WMMA scores -------------------------------------------
// scores[row] = sum_n v[n] * tanh( enc[row]@W1[:,n] + W1b[n] + dec_proj[b][n] )
__global__ __launch_bounds__(256) void scores_kernel(const float* __restrict__ enc,
                                                     const unsigned short* __restrict__ w1t,
                                                     const float* __restrict__ w1b,
                                                     const float* __restrict__ vw,
                                                     const float* __restrict__ decp,
                                                     float* __restrict__ scores) {
    extern __shared__ char smem[];
    unsigned short* Abf = (unsigned short*)smem;                 // 64 x 1032 bf16 = 132096 B
    float* bias = (float*)(smem + (size_t)MBLK * LDSK * 2);      // 1024 f32
    float* vls  = bias + HD;                                     // 1024 f32
    float* sacc = vls + HD;                                      // 64 f32

    const int tid     = threadIdx.x;
    const int rowbase = blockIdx.x * MBLK;        // flattened row in [B*S]
    const int b       = rowbase / SS;

    for (int i = tid; i < HD; i += 256) {
        bias[i] = w1b[i] + decp[(size_t)b * HD + i];
        vls[i]  = vw[i];
    }
    if (tid < MBLK) sacc[tid] = 0.f;

    // Stage A tile: 64 x 1024 fp32 -> bf16 in LDS (coalesced float4 loads)
    const float4* encv = (const float4*)(enc + (size_t)rowbase * HD);
#pragma unroll 4
    for (int j = 0; j < (MBLK * HD / 4) / 256; ++j) {   // 64 iterations
        const int idx = tid + j * 256;                  // float4 index
        const int row = idx >> 8;                       // 256 float4 per row
        const int k   = (idx & 255) * 4;
        float4 f = encv[idx];
        unsigned u0 = ((unsigned)f2bf(f.y) << 16) | f2bf(f.x);
        unsigned u1 = ((unsigned)f2bf(f.w) << 16) | f2bf(f.z);
        *(uint2*)(Abf + (size_t)row * LDSK + k) = make_uint2(u0, u1);
    }
    __syncthreads();

    const int wave  = tid >> 5;
    const int lane  = tid & 31;
    const int mwave = wave & 1;      // 2 M-waves  x 2 M-tiles  -> M = 64
    const int nwave = wave >> 1;     // 4 N-waves  x 2 N-tiles  -> Nchunk = 128
    const int half  = lane >> 4;
    const int l     = lane & 15;

    // per-lane fragment base pointers (K element offset added per step)
    const unsigned short* ap[2];
#pragma unroll
    for (int mt = 0; mt < 2; ++mt)
        ap[mt] = Abf + (size_t)(mwave * 32 + mt * 16 + l) * LDSK + half * 8;

    float sp[2][8];
#pragma unroll
    for (int mt = 0; mt < 2; ++mt)
#pragma unroll
        for (int i = 0; i < 8; ++i) sp[mt][i] = 0.f;

    for (int nc = 0; nc < HD / 128; ++nc) {
        const int colb0 = nc * 128 + nwave * 32;
        const unsigned short* bp[2];
#pragma unroll
        for (int nt = 0; nt < 2; ++nt)
            bp[nt] = w1t + (size_t)(colb0 + nt * 16 + l) * HD + half * 8;

        v8f acc[2][2] = {};
        v16bf am[2][2], bm[2][2];   // [pipeline buffer][tile]

        auto lda = [&](int buf, int kk) {
#pragma unroll
            for (int mt = 0; mt < 2; ++mt) {
                const uint4* p = (const uint4*)(ap[mt] + kk);
                union { uint4 u[2]; v16bf v; } t;
                t.u[0] = p[0];        // K = kk .. kk+7
                t.u[1] = p[2];        // K = kk+16 .. kk+23
                am[buf][mt] = t.v;
            }
        };
        auto ldb = [&](int buf, int kk) {
#pragma unroll
            for (int nt = 0; nt < 2; ++nt) {
                const uint4* q = (const uint4*)(bp[nt] + kk);
                union { uint4 u[2]; v16bf v; } t;
                t.u[0] = q[0];
                t.u[1] = q[2];
                bm[buf][nt] = t.v;
            }
        };
        auto domma = [&](int buf) {
#pragma unroll
            for (int mt = 0; mt < 2; ++mt)
#pragma unroll
                for (int nt = 0; nt < 2; ++nt)
                    acc[mt][nt] = __builtin_amdgcn_wmma_f32_16x16x32_bf16(
                        false, am[buf][mt], false, bm[buf][nt], (short)0,
                        acc[mt][nt], false, false);
        };

        // 2-deep software pipeline over the 32 K-steps
        lda(0, 0); ldb(0, 0);
#pragma unroll 2
        for (int ks = 0; ks < HD / 32; ks += 2) {
            const int kk1 = ((ks + 1) & 31) * 32;
            lda(1, kk1); ldb(1, kk1);
            domma(0);
            const int kk2 = ((ks + 2) & 31) * 32;   // tail wraps: harmless discarded prefetch
            lda(0, kk2); ldb(0, kk2);
            domma(1);
        }

        // fused epilogue: +bias, tanh, *v, accumulate per-row score partials
#pragma unroll
        for (int nt = 0; nt < 2; ++nt) {
            const int ci = colb0 + nt * 16 + l;
            const float bb = bias[ci];
            const float vv = vls[ci];
#pragma unroll
            for (int mt = 0; mt < 2; ++mt)
#pragma unroll
                for (int i = 0; i < 8; ++i)
                    sp[mt][i] = fmaf(vv, TANHF(acc[mt][nt][i] + bb), sp[mt][i]);
        }
    }
    // per-row reduction across lanes/waves via LDS float atomics
#pragma unroll
    for (int mt = 0; mt < 2; ++mt)
#pragma unroll
        for (int i = 0; i < 8; ++i)
            atomicAdd(&sacc[mwave * 32 + mt * 16 + half * 8 + i], sp[mt][i]);
    __syncthreads();
    if (tid < MBLK) scores[(size_t)rowbase + tid] = sacc[tid];
}

// ---------------- Kernel 4: softmax over S, in place ------------------------------------
__global__ __launch_bounds__(256) void softmax_kernel(float* __restrict__ attn) {
    __shared__ float red[256];
    const int tid = threadIdx.x;
    float* row = attn + (size_t)blockIdx.x * SS;
    float v[SS / 256];
    float m = -INFINITY;
#pragma unroll
    for (int i = 0; i < SS / 256; ++i) {
        v[i] = row[tid + i * 256];
        m = fmaxf(m, v[i]);
    }
    red[tid] = m;
    __syncthreads();
    for (int s = 128; s > 0; s >>= 1) {
        if (tid < s) red[tid] = fmaxf(red[tid], red[tid + s]);
        __syncthreads();
    }
    m = red[0];
    __syncthreads();
    float sum = 0.f;
#pragma unroll
    for (int i = 0; i < SS / 256; ++i) {
        v[i] = __expf(v[i] - m);
        sum += v[i];
    }
    red[tid] = sum;
    __syncthreads();
    for (int s = 128; s > 0; s >>= 1) {
        if (tid < s) red[tid] += red[tid + s];
        __syncthreads();
    }
    const float inv = 1.0f / red[0];
#pragma unroll
    for (int i = 0; i < SS / 256; ++i)
        row[tid + i * 256] = v[i] * inv;
}

// ---------------- Kernel 5: context[b][h] = sum_s w[b][s] * enc[b][s][h] ----------------
__global__ __launch_bounds__(256) void context_kernel(const float* __restrict__ enc,
                                                      const float* __restrict__ attn,
                                                      float* __restrict__ ctx) {
    __shared__ float w[SS];
    const int b   = blockIdx.y;
    const int h   = blockIdx.x * 256 + threadIdx.x;
    for (int i = threadIdx.x; i < SS; i += 256)
        w[i] = attn[(size_t)b * SS + i];
    __syncthreads();
    const float* e = enc + (size_t)b * SS * HD + h;
    float a0 = 0.f, a1 = 0.f, a2 = 0.f, a3 = 0.f;
#pragma unroll 4
    for (int s = 0; s < SS; s += 4) {
        a0 = fmaf(w[s + 0], e[(size_t)(s + 0) * HD], a0);
        a1 = fmaf(w[s + 1], e[(size_t)(s + 1) * HD], a1);
        a2 = fmaf(w[s + 2], e[(size_t)(s + 2) * HD], a2);
        a3 = fmaf(w[s + 3], e[(size_t)(s + 3) * HD], a3);
    }
    ctx[(size_t)b * HD + h] = (a0 + a1) + (a2 + a3);
}

extern "C" void kernel_launch(void* const* d_in, const int* in_sizes, int n_in,
                              void* d_out, int out_size, void* d_ws, size_t ws_size,
                              hipStream_t stream) {
    (void)in_sizes; (void)n_in; (void)out_size; (void)ws_size;
    const float* dec_hidden = (const float*)d_in[0];   // [B,H]
    const float* enc        = (const float*)d_in[1];   // [B,S,H]
    const float* W1_w       = (const float*)d_in[2];   // [H,H]
    const float* W1_b       = (const float*)d_in[3];   // [H]
    const float* W2_w       = (const float*)d_in[4];   // [H,H]
    const float* W2_b       = (const float*)d_in[5];   // [H]
    const float* v_w        = (const float*)d_in[6];   // [H]

    float* ctx  = (float*)d_out;                       // [B,H]
    float* attn = (float*)d_out + (size_t)BB * HD;     // [B,S] (scores -> softmax in place)

    unsigned short* w1t = (unsigned short*)d_ws;                     // H*H bf16 = 2 MB
    float* decp = (float*)((char*)d_ws + (size_t)HD * HD * 2);       // B*H f32

    const int smem_bytes = MBLK * LDSK * 2 + 2 * HD * 4 + MBLK * 4;  // 140544
    (void)hipFuncSetAttribute((const void*)scores_kernel,
                              hipFuncAttributeMaxDynamicSharedMemorySize, smem_bytes);

    w1t_kernel<<<dim3(HD / 32, HD / 32), 256, 0, stream>>>(W1_w, w1t);
    decproj_kernel<<<(BB * HD) / 256, 256, 0, stream>>>(dec_hidden, W2_w, W2_b, decp);
    scores_kernel<<<(BB * SS) / MBLK, 256, smem_bytes, stream>>>(enc, w1t, W1_b, v_w, decp, attn);
    softmax_kernel<<<BB, 256, 0, stream>>>(attn);
    context_kernel<<<dim3(HD / 256, BB), 256, 0, stream>>>(enc, attn, ctx);
}
